// MultiScaleRetention_28741921145323
// MI455X (gfx1250) — matrix operand order repaired and checked by reading, not compiled, not run on hardware
//
#include <hip/hip_runtime.h>
#include <hip/hip_bf16.h>

typedef _Float16 h16;
typedef __attribute__((ext_vector_type(16))) _Float16 v16h;
typedef __attribute__((ext_vector_type(8)))  float    v8f;
typedef __attribute__((ext_vector_type(4)))  int      v4i;

#define V8F_ZERO {0.f,0.f,0.f,0.f,0.f,0.f,0.f,0.f}

// ---------------- async global->LDS copy (gfx1250 async-DMA path) ----------------
#if defined(__has_builtin)
#if __has_builtin(__builtin_amdgcn_global_load_async_to_lds_b128) && \
    __has_builtin(__builtin_amdgcn_s_wait_asynccnt)
#define USE_ASYNC 1
#endif
#endif
#ifndef USE_ASYNC
#define USE_ASYNC 0
#endif

#if USE_ASYNC
typedef __attribute__((address_space(1))) v4i gv4i;   // global int4
typedef __attribute__((address_space(3))) v4i lv4i;   // LDS int4
#endif

__device__ __forceinline__ void cp_b128(void* lds_dst, const void* gsrc) {
#if USE_ASYNC
  __builtin_amdgcn_global_load_async_to_lds_b128(
      (gv4i*)(unsigned long long)gsrc,
      (lv4i*)(unsigned int)(unsigned long long)lds_dst,
      0, 0);
#else
  *(uint4*)lds_dst = *(const uint4*)gsrc;
#endif
}
#if USE_ASYNC
#define CP_WAIT(n) __builtin_amdgcn_s_wait_asynccnt(n)
#else
#define CP_WAIT(n) ((void)0)
#endif

// ---------------- fragment loaders (CDNA5 wave32 WMMA 16x16x32 f16) ----------------
// A (16x32, MxK): lane<16 -> row=lane, K {0..7,16..23}; lane>=16 -> row=lane-16, K {8..15,24..31}
__device__ __forceinline__ v16h load_a_frag(const h16* base, int ldh) {
  int lane = threadIdx.x & 31;
  int row  = lane & 15;
  int hs   = lane >> 4;
  union { v16h h; unsigned int u[8]; } f;
#pragma unroll
  for (int v = 0; v < 8; ++v) {
    int kb = ((v < 4) ? 2 * v : 16 + 2 * (v - 4)) + hs * 8;
    f.u[v] = *(const unsigned int*)(base + row * ldh + kb);
  }
  return f.h;
}
// B (32x16, KxN): base points at [n][k] (n-major) storage, ldh = k-stride
__device__ __forceinline__ v16h load_b_frag(const h16* base, int ldh) {
  int lane = threadIdx.x & 31;
  int n    = lane & 15;
  int kh   = lane >> 4;
  union { v16h h; unsigned int u[8]; } f;
#pragma unroll
  for (int v = 0; v < 8; ++v) {
    int k = ((v < 4) ? 2 * v : 16 + 2 * (v - 4)) + kh * 8;
    f.u[v] = *(const unsigned int*)(base + n * ldh + k);
  }
  return f.h;
}

// ---------------- f32 -> f16 convert ----------------
__global__ void msr_cvt_f16(const float* __restrict__ src, h16* __restrict__ dst, int n) {
  int i = blockIdx.x * blockDim.x + threadIdx.x;
  if (i < n) dst[i] = (h16)src[i];
}
// convert + transpose: src K x N (row-major) -> dst N x K (row-major)
__global__ void msr_cvt_f16_T(const float* __restrict__ src, h16* __restrict__ dst,
                              int K, int N) {
  int idx = blockIdx.x * blockDim.x + threadIdx.x;
  if (idx >= K * N) return;
  int n = idx / K, k = idx - n * K;          // dst writes coalesced along k
  dst[idx] = (h16)src[(size_t)k * N + n];
}

// ---------------- tiled WMMA GEMM: C = (A @ B + bias) * scale ----------------
// A: MxK f16 row-major, Bt: NxK f16 (pre-transposed), C: MxN f32.
// Tiles 128x128x32, 256 thr (8 waves), double-buffered async LDS staging.
__global__ __launch_bounds__(256)
void msr_gemm_wmma(const h16* __restrict__ A, const h16* __restrict__ Bt,
                   const float* __restrict__ bias, float* __restrict__ C,
                   int M, int N, int K, float scale) {
  __shared__ h16 As[2][128 * 32];
  __shared__ h16 Bs[2][128 * 32];   // [n][k]
  const int tid  = threadIdx.x;
  const int wave = tid >> 5;
  const int wm   = wave >> 2;       // 0..1
  const int wn   = wave & 3;        // 0..3
  const int m0   = blockIdx.y * 128;
  const int n0   = blockIdx.x * 128;

  v8f acc[4][2];
#pragma unroll
  for (int i = 0; i < 4; ++i)
#pragma unroll
    for (int j = 0; j < 2; ++j) { v8f z = V8F_ZERO; acc[i][j] = z; }

  auto stage = [&](int buf, int kk) {
#pragma unroll
    for (int i = 0; i < 2; ++i) {
      int id  = tid * 2 + i;        // 0..511
      int row = id >> 2;            // 0..127
      int kp  = (id & 3) * 8;
      cp_b128(&As[buf][row * 32 + kp], A  + (size_t)(m0 + row) * K + kk + kp);
      cp_b128(&Bs[buf][row * 32 + kp], Bt + (size_t)(n0 + row) * K + kk + kp);
    }
  };

  const int nk = K >> 5;
  stage(0, 0);
  for (int ik = 0; ik < nk; ++ik) {
    const int cur = ik & 1;
    if (ik + 1 < nk) { stage(cur ^ 1, (ik + 1) * 32); CP_WAIT(4); }
    else             { CP_WAIT(0); }
    __syncthreads();

    v16h bfr[2];
#pragma unroll
    for (int nt = 0; nt < 2; ++nt)
      bfr[nt] = load_b_frag(&Bs[cur][(wn * 32 + nt * 16) * 32], 32);
#pragma unroll
    for (int mt = 0; mt < 4; ++mt) {
      v16h af = load_a_frag(&As[cur][(wm * 64 + mt * 16) * 32], 32);
#pragma unroll
      for (int nt = 0; nt < 2; ++nt)
        acc[mt][nt] = __builtin_amdgcn_wmma_f32_16x16x32_f16(
            false, af, false, bfr[nt], (short)0, acc[mt][nt], false, false);
    }
    __syncthreads();
  }

  const int lane = tid & 31, ln = lane & 15, lh = lane >> 4;
#pragma unroll
  for (int nt = 0; nt < 2; ++nt) {
    int col  = n0 + wn * 32 + nt * 16 + ln;
    float bv = bias[col];
#pragma unroll
    for (int mt = 0; mt < 4; ++mt)
#pragma unroll
      for (int r = 0; r < 8; ++r) {
        int row = m0 + wm * 64 + mt * 16 + lh * 8 + r;
        C[(size_t)row * N + col] = (acc[mt][nt][r] + bv) * scale;
      }
  }
}

// ---------------- RoPE: q(B,S,1024) f32 -> qr(B,H,S,64) f16 ----------------
__global__ void msr_rope(const float* __restrict__ q, h16* __restrict__ out, int B, int S) {
  int idx = blockIdx.x * blockDim.x + threadIdx.x;
  if (idx >= B * S * 1024) return;
  int c = idx & 1023;
  int s = (idx >> 10) % S;
  int b = idx / (S * 1024);
  int h = c >> 6, d = c & 63;
  float cur = q[idx];
  float oth = (d & 1) ? q[idx - 1] : q[idx + 1];
  float rot = (d & 1) ? oth : -oth;
  float ang = __expf(-((float)(d >> 1) / 31.0f) * 9.210340371976184f); // 10000^-lin
  float th  = (float)s * ang;
  float val = cur * __cosf(th) + rot * __sinf(th);
  out[(((size_t)b * 16 + h) * S + s) * 64 + d] = (h16)val;
}

// ---------------- v(B,S,2048) f32 -> vrT(B,H,128,S) f16 (e-major, pre-transposed) --------
__global__ void msr_reshape_vT(const float* __restrict__ v, h16* __restrict__ vrT, int B, int S) {
  int idx = blockIdx.x * blockDim.x + threadIdx.x;
  if (idx >= B * S * 2048) return;
  int c = idx & 2047;
  int s = (idx >> 11) % S;
  int b = idx / (S * 2048);
  int h = c >> 7, e = c & 127;
  vrT[((((size_t)b * 16 + h) * 128) + e) * S + s] = (h16)v[idx];
}

// ---------------- fused retention attention + LayerNorm ----------------
// grid: (S/128, B*H); block 256 (8 waves, 16 rows each). out: (B,S,2048) f32 at col h*128+e.
__global__ __launch_bounds__(256)
void msr_attention(const h16* __restrict__ qr, const h16* __restrict__ kr,
                   const h16* __restrict__ vrT, float* __restrict__ out, int S) {
  __shared__ h16 krLds[32 * 64];    // [t][d] natural
  __shared__ h16 vrLds[128 * 32];   // [e][t]
  __shared__ h16 sLds[8 * 16 * 32]; // per-wave masked-score tile (A-matrix 16x32)

  const int tid  = threadIdx.x;
  const int wave = tid >> 5;
  const int lane = tid & 31, ln = lane & 15, lh = lane >> 4;
  const int h  = blockIdx.y & 15;
  const int b  = blockIdx.y >> 4;
  const size_t bh = (size_t)b * 16 + h;
  const int s0 = blockIdx.x * 128;
  const int srow0 = s0 + wave * 16;

  const float e_decay = 1.0f - exp2f(-5.0f - (float)h);
  const float decay   = logf(e_decay);
  const float inv_1md = 1.0f / (1.0f - e_decay);

  float cs[8];                        // 1/sqrt(mask row sum), closed form
#pragma unroll
  for (int r = 0; r < 8; ++r) {
    int srow = srow0 + lh * 8 + r;
    float rsum = (1.0f - __expf(decay * (float)(srow + 1))) * inv_1md;
    cs[r] = rsqrtf(rsum);
  }

  const h16* qbase = qr + (bh * S + srow0) * 64;
  v16h aq0 = load_a_frag(qbase, 64);
  v16h aq1 = load_a_frag(qbase + 32, 64);

  float rs[8];
#pragma unroll
  for (int r = 0; r < 8; ++r) rs[r] = 0.0f;
  v8f accO[8];
#pragma unroll
  for (int nt = 0; nt < 8; ++nt) { v8f z = V8F_ZERO; accO[nt] = z; }

  h16* sW = &sLds[wave * 512];
  const int tend = s0 + 128;
  for (int t0 = 0; t0 < tend; t0 += 32) {
    // stage kr tile 32x64 (1 async b128 per thread)
    {
      int tr = tid >> 3, dp = (tid & 7) * 8;
      cp_b128(&krLds[tr * 64 + dp], kr + (bh * S + t0 + tr) * 64 + dp);
    }
    // stage vrT tile 128x32 (2 async b128 per thread, already e-major)
#pragma unroll
    for (int i = 0; i < 2; ++i) {
      int id = tid * 2 + i;
      int e = id >> 2, tp = (id & 3) * 8;
      cp_b128(&vrLds[e * 32 + tp], vrT + (bh * 128 + e) * (size_t)S + t0 + tp);
    }
    CP_WAIT(0);
    __syncthreads();

    const bool active = (t0 <= srow0 + 15);
    if (active) {
#pragma unroll
      for (int tt = 0; tt < 32; tt += 16) {
        v16h bk0 = load_b_frag(&krLds[tt * 64 + 0], 64);
        v16h bk1 = load_b_frag(&krLds[tt * 64 + 32], 64);
        v8f sc = V8F_ZERO;
        sc = __builtin_amdgcn_wmma_f32_16x16x32_f16(false, aq0, false, bk0, (short)0, sc, false, false);
        sc = __builtin_amdgcn_wmma_f32_16x16x32_f16(false, aq1, false, bk1, (short)0, sc, false, false);
        int tcol = t0 + tt + ln;
#pragma unroll
        for (int r = 0; r < 8; ++r) {
          int srow = srow0 + lh * 8 + r;
          float val = sc[r];
          val = (tcol <= srow) ? val * __expf(decay * (float)(srow - tcol)) * cs[r] : 0.0f;
          rs[r] += val;
          sW[(lh * 8 + r) * 32 + tt + ln] = (h16)val;
        }
      }
    }
    __syncthreads();
    if (active) {
      v16h as = load_a_frag(sW, 32);
#pragma unroll
      for (int nt = 0; nt < 8; ++nt) {
        v16h bv = load_b_frag(&vrLds[nt * 16 * 32], 32);
        accO[nt] = __builtin_amdgcn_wmma_f32_16x16x32_f16(
            false, as, false, bv, (short)0, accO[nt], false, false);
      }
    }
    __syncthreads();
  }

  // denom = clip(|row sum|, 1): reduce over the 16 lanes owning each row
#pragma unroll
  for (int r = 0; r < 8; ++r) {
    float v = rs[r];
    v += __shfl_xor(v, 1, 32);
    v += __shfl_xor(v, 2, 32);
    v += __shfl_xor(v, 4, 32);
    v += __shfl_xor(v, 8, 32);
    float inv = 1.0f / fmaxf(1.0f, fabsf(v));
#pragma unroll
    for (int nt = 0; nt < 8; ++nt) accO[nt][r] *= inv;
  }

  // LayerNorm over 128 head dims per row, scatter to (B,S,2048)
#pragma unroll
  for (int r = 0; r < 8; ++r) {
    float sum = 0.f, sq = 0.f;
#pragma unroll
    for (int nt = 0; nt < 8; ++nt) { float v = accO[nt][r]; sum += v; sq += v * v; }
    sum += __shfl_xor(sum, 1, 32); sq += __shfl_xor(sq, 1, 32);
    sum += __shfl_xor(sum, 2, 32); sq += __shfl_xor(sq, 2, 32);
    sum += __shfl_xor(sum, 4, 32); sq += __shfl_xor(sq, 4, 32);
    sum += __shfl_xor(sum, 8, 32); sq += __shfl_xor(sq, 8, 32);
    float mu   = sum * (1.0f / 128.0f);
    float var  = sq * (1.0f / 128.0f) - mu * mu;
    float rstd = rsqrtf(var + 1e-5f);
    int srow = srow0 + lh * 8 + r;
    size_t rowoff = ((size_t)b * S + srow) * 2048 + h * 128;
#pragma unroll
    for (int nt = 0; nt < 8; ++nt)
      out[rowoff + nt * 16 + ln] = (accO[nt][r] - mu) * rstd;
  }
}

// ---------------- gate: ah = f16( silu(g) * attn ) ----------------
__global__ void msr_gate(const float* __restrict__ g, const float* __restrict__ attn,
                         h16* __restrict__ ah, int n) {
  int i = blockIdx.x * blockDim.x + threadIdx.x;
  if (i >= n) return;
  float gv = g[i];
  float si = gv / (1.0f + __expf(-gv));
  ah[i] = (h16)(si * attn[i]);
}

// ---------------- host-side orchestration ----------------
#define MB(x) ((size_t)(x) * 1024u * 1024u)

extern "C" void kernel_launch(void* const* d_in, const int* in_sizes, int n_in,
                              void* d_out, int out_size, void* d_ws, size_t ws_size,
                              hipStream_t stream) {
  const float* x  = (const float*)d_in[0];
  const float* Wq = (const float*)d_in[1];
  const float* bq = (const float*)d_in[2];
  const float* Wk = (const float*)d_in[3];
  const float* bk = (const float*)d_in[4];
  const float* Wv = (const float*)d_in[5];
  const float* bv = (const float*)d_in[6];
  const float* Wg = (const float*)d_in[7];
  const float* bg = (const float*)d_in[8];
  const float* Wo = (const float*)d_in[9];
  const float* bo = (const float*)d_in[10];

  const int B = 2, S = 2048, D = 1024, M = B * S;   // 4096
  const float kscale = 0.125f;                      // 64^-0.5

  char* ws = (char*)d_ws;
  h16*   xh  = (h16*)(ws + MB(0));     // 8 MB
  h16*   Wh  = (h16*)(ws + MB(8));     // 4 MB (reused per weight, N x K layout)
  float* qf  = (float*)(ws + MB(12));  // 16 MB
  float* kf  = (float*)(ws + MB(28));  // 16 MB
  float* vf  = (float*)(ws + MB(44));  // 32 MB
  float* gf  = (float*)(ws + MB(76));  // 32 MB
  h16*   qrp = (h16*)(ws + MB(108));   // 8 MB
  h16*   krp = (h16*)(ws + MB(116));   // 8 MB
  h16*   vrT = (h16*)(ws + MB(124));   // 16 MB
  float* attn = (float*)(ws + MB(12)); // 32 MB, reuses qf+kf (free after RoPE)
  h16*   ah   = (h16*)(ws + MB(44));   // 16 MB, reuses vf (free after reshape)

  const int T = 256;
  msr_cvt_f16<<<(M * D + T - 1) / T, T, 0, stream>>>(x, xh, M * D);

  // q = x@Wq + bq
  msr_cvt_f16_T<<<(D * D + T - 1) / T, T, 0, stream>>>(Wq, Wh, D, D);
  msr_gemm_wmma<<<dim3(D / 128, M / 128), T, 0, stream>>>(xh, Wh, bq, qf, M, D, D, 1.0f);
  // k = (x@Wk + bk) * scaling
  msr_cvt_f16_T<<<(D * D + T - 1) / T, T, 0, stream>>>(Wk, Wh, D, D);
  msr_gemm_wmma<<<dim3(D / 128, M / 128), T, 0, stream>>>(xh, Wh, bk, kf, M, D, D, kscale);
  // v = x@Wv + bv
  msr_cvt_f16_T<<<(D * 2 * D + T - 1) / T, T, 0, stream>>>(Wv, Wh, D, 2 * D);
  msr_gemm_wmma<<<dim3(2 * D / 128, M / 128), T, 0, stream>>>(xh, Wh, bv, vf, M, 2 * D, D, 1.0f);
  // g = x@Wg + bg
  msr_cvt_f16_T<<<(D * 2 * D + T - 1) / T, T, 0, stream>>>(Wg, Wh, D, 2 * D);
  msr_gemm_wmma<<<dim3(2 * D / 128, M / 128), T, 0, stream>>>(xh, Wh, bg, gf, M, 2 * D, D, 1.0f);

  // RoPE + head-major reshapes (f16)
  msr_rope<<<(B * S * D + T - 1) / T, T, 0, stream>>>(qf, qrp, B, S);
  msr_rope<<<(B * S * D + T - 1) / T, T, 0, stream>>>(kf, krp, B, S);
  msr_reshape_vT<<<(B * S * 2 * D + T - 1) / T, T, 0, stream>>>(vf, vrT, B, S);

  // fused retention attention + LayerNorm -> attn (B,S,2048)
  msr_attention<<<dim3(S / 128, B * 16), T, 0, stream>>>(qrp, krp, vrT, attn, S);

  // gated activation -> f16
  msr_gate<<<(B * S * 2 * D + T - 1) / T, T, 0, stream>>>(gf, attn, ah, B * S * 2 * D);

  // out = ah @ Wo + bo  (f32 output)
  msr_cvt_f16_T<<<(2 * D * D + T - 1) / T, T, 0, stream>>>(Wo, Wh, 2 * D, D);
  msr_gemm_wmma<<<dim3(D / 128, M / 128), T, 0, stream>>>(ah, Wh, bo, (float*)d_out,
                                                          M, D, 2 * D, 1.0f);
}
